// FocusedAttnV2_65859028517418
// MI455X (gfx1250) — matrix operand order repaired and checked by reading, not compiled, not run on hardware
//
#include <hip/hip_runtime.h>
#include <hip/hip_bf16.h>

// ---------------------------------------------------------------------------
// FocusedAttnV2 for MI455X (gfx1250, wave32, WMMA 16x16x32 bf16)
// B=4, N_Q=256, N_KV=8192, C=256, H=8, head_dim=32
// Memory-bound (~70MB in, ~17.4 GFLOP): f32 -> bf16 once, intermediates
// L2-resident, all matmuls on v_wmma, K/V staged to LDS with
// global_load_async_to_lds_b128 (ASYNCcnt) and double buffering.
// ---------------------------------------------------------------------------

typedef __attribute__((ext_vector_type(16))) __bf16 v16bf;
typedef __attribute__((ext_vector_type(8)))  float  v8f;

struct alignas(16) Raw16  { unsigned int w[4]; };
struct alignas(16) Raw32  { unsigned int w[8]; };
struct alignas(16) Raw16x2 { Raw16 a, b; };

// 16 contiguous bf16 -> B fragment (lane-local part)
static __device__ inline v16bf ld_bf16x16(const __bf16* p) {
  Raw32 r = *(const Raw32*)p;
  return __builtin_bit_cast(v16bf, r);
}
// two chunks of 8 contiguous bf16 -> A fragment (lane-local part)
static __device__ inline v16bf ld_bf16x8x2(const __bf16* p0, const __bf16* p1) {
  Raw16x2 t;
  t.a = *(const Raw16*)p0;
  t.b = *(const Raw16*)p1;
  return __builtin_bit_cast(v16bf, t);
}
// 16 contiguous f32 -> bf16 B fragment
static __device__ inline v16bf cvt_f32x16(const float* p) {
  float4 a = *(const float4*)(p + 0);
  float4 b = *(const float4*)(p + 4);
  float4 c = *(const float4*)(p + 8);
  float4 d = *(const float4*)(p + 12);
  v16bf r;
  r[0]=(__bf16)a.x;  r[1]=(__bf16)a.y;  r[2]=(__bf16)a.z;  r[3]=(__bf16)a.w;
  r[4]=(__bf16)b.x;  r[5]=(__bf16)b.y;  r[6]=(__bf16)b.z;  r[7]=(__bf16)b.w;
  r[8]=(__bf16)c.x;  r[9]=(__bf16)c.y;  r[10]=(__bf16)c.z; r[11]=(__bf16)c.w;
  r[12]=(__bf16)d.x; r[13]=(__bf16)d.y; r[14]=(__bf16)d.z; r[15]=(__bf16)d.w;
  return r;
}
// two chunks of 8 contiguous f32 -> bf16 A fragment
static __device__ inline v16bf cvt_f32x8x2(const float* p0, const float* p1) {
  float4 a = *(const float4*)(p0 + 0);
  float4 b = *(const float4*)(p0 + 4);
  float4 c = *(const float4*)(p1 + 0);
  float4 d = *(const float4*)(p1 + 4);
  v16bf r;
  r[0]=(__bf16)a.x;  r[1]=(__bf16)a.y;  r[2]=(__bf16)a.z;  r[3]=(__bf16)a.w;
  r[4]=(__bf16)b.x;  r[5]=(__bf16)b.y;  r[6]=(__bf16)b.z;  r[7]=(__bf16)b.w;
  r[8]=(__bf16)c.x;  r[9]=(__bf16)c.y;  r[10]=(__bf16)c.z; r[11]=(__bf16)c.w;
  r[12]=(__bf16)d.x; r[13]=(__bf16)d.y; r[14]=(__bf16)d.z; r[15]=(__bf16)d.w;
  return r;
}

#define WMMA_BF16(A, B, C) \
  __builtin_amdgcn_wmma_f32_16x16x32_bf16(false, (A), false, (B), (short)0, (C), false, false)

// Async global->LDS copy, 16 bytes per lane (ASYNCcnt-tracked).
// VDST = LDS byte address, VADDR = 64-bit global address (GV mode).
static __device__ inline void async_lds_b128(unsigned ldsoff, const void* g) {
  asm volatile("global_load_async_to_lds_b128 %0, %1, off"
               : : "v"(ldsoff), "v"((unsigned long long)(size_t)g)
               : "memory");
}
static __device__ inline void wait_async0() {
  asm volatile("s_wait_asynccnt 0" : : : "memory");
}

#define BQ   4
#define NQ   256
#define NKV  8192
#define CC   256
#define HH   8
#define HD   32
#define QSCALE 0.17677669529663689f   // 1/sqrt(32)
#define LROW 40                       // LDS row stride in bf16 (80B, conflict-free)

// ---------------------------------------------------------------------------
// Stage A: projections.  grid.x = row strips of 16, grid.y = job
//   job 0: KH  = k @ Wk^T                 -> bf16 row-major   (B*NKV, C)
//   job 1: VHT = (v @ Wv^T) transposed    -> bf16 d-major     [b*C + c][kv]
//   job 2: QH  = (q @ Wk^T) * scale       -> bf16 row-major   (B*NQ, C)
// block = 128 threads = 4 waves, each wave computes 16 rows x 64 cols.
// ---------------------------------------------------------------------------
__global__ __launch_bounds__(128) void ProjKernel(
    const float* __restrict__ q, const float* __restrict__ k,
    const float* __restrict__ v, const float* __restrict__ Wk,
    const float* __restrict__ Wv,
    __bf16* __restrict__ KH, __bf16* __restrict__ VHT, __bf16* __restrict__ QH)
{
  const int job = blockIdx.y;
  const float* src; const float* W; int M;
  if (job == 0)      { src = k; W = Wk; M = BQ * NKV; }
  else if (job == 1) { src = v; W = Wv; M = BQ * NKV; }
  else               { src = q; W = Wk; M = BQ * NQ;  }

  const int m0 = blockIdx.x * 16;
  if (m0 >= M) return;

  const int lane = threadIdx.x & 31;
  const int wib  = threadIdx.x >> 5;
  const int lo   = lane & 15;
  const int hi   = lane >> 4;
  const int n00  = wib * 64;

  v8f acc[4] = {};

  const float* arow = src + (size_t)(m0 + lo) * CC;
  #pragma unroll
  for (int kt = 0; kt < 8; ++kt) {
    v16bf afrag = cvt_f32x16(arow + kt * 32 + hi * 16);
    #pragma unroll
    for (int nt = 0; nt < 4; ++nt) {
      const float* wrow = W + (size_t)(n00 + nt * 16 + lo) * CC + kt * 32 + hi * 16;
      v16bf bfrag = cvt_f32x16(wrow);
      acc[nt] = WMMA_BF16(afrag, bfrag, acc[nt]);
    }
  }

  const float scale = (job == 2) ? QSCALE : 1.0f;
  #pragma unroll
  for (int nt = 0; nt < 4; ++nt) {
    const int col = n00 + nt * 16 + lo;
    #pragma unroll
    for (int vv = 0; vv < 8; ++vv) {
      const int r = m0 + vv + 8 * hi;
      const float val = acc[nt][vv] * scale;
      if (job == 0) {
        KH[(size_t)r * CC + col] = (__bf16)val;
      } else if (job == 1) {
        const int b  = r >> 13;           // r / 8192
        const int kv = r & (NKV - 1);
        VHT[(size_t)(b * CC + col) * NKV + kv] = (__bf16)val;
      } else {
        QH[(size_t)r * CC + col] = (__bf16)val;
      }
    }
  }
}

// ---------------------------------------------------------------------------
// Stage B: flash attention. One block per (b, h): 16 waves, one 16-row q tile
// each. 32-kv K/V chunks staged into double-buffered LDS once per block via
// global_load_async_to_lds_b128 (waves 0/1 are producers), fenced with
// s_wait_asynccnt + barrier. All waves consume via ds_load_b128 fragments:
// 2 score WMMAs, mask, online softmax, P transpose through LDS, 2 PV WMMAs.
// ---------------------------------------------------------------------------
__global__ __launch_bounds__(512) void AttnKernel(
    const __bf16* __restrict__ QH, const __bf16* __restrict__ KH,
    const __bf16* __restrict__ VHT, const unsigned char* __restrict__ mask,
    float* __restrict__ X)
{
  // K chunk: [kv_local 0..31][d 0..31], padded rows; V chunk: [d 0..31][kv_local]
  __shared__ __align__(16) __bf16 Klds[2][32 * LROW];
  __shared__ __align__(16) __bf16 Vlds[2][32 * LROW];
  __shared__ __align__(16) __bf16 Plds[16][16 * 32];

  const int lane = threadIdx.x & 31;
  const int wib  = threadIdx.x >> 5;          // 0..15 = q tile
  const int bb   = blockIdx.x >> 3;           // batch
  const int h    = blockIdx.x & 7;            // head
  const int lo   = lane & 15;
  const int hi   = lane >> 4;
  const int qBase = wib * 16;

  // Q A-fragment (held for the whole kv sweep)
  const __bf16* qrow = QH + (size_t)(bb * NQ + qBase + lo) * CC + h * HD;
  const v16bf qf = ld_bf16x16(qrow + hi * 16);

  v8f o0 = {}, o1 = {};
  float run_m[8], run_s[8];
  #pragma unroll
  for (int vv = 0; vv < 8; ++vv) { run_m[vv] = -1e30f; run_s[vv] = 0.0f; }

  // mask base: row (qBase + vv + 8*hi), offsets vv*NKV folded into immediates
  const unsigned char* mbase = mask + (size_t)(qBase + 8 * hi) * NKV + lo;

  // producer: wave 0 stages K chunk, wave 1 stages V chunk (2KB each = 4 x b128/lane)
  auto issue_chunk = [&](int buf, int kv0n) {
    if (wib == 0) {
      #pragma unroll
      for (int j = 0; j < 4; ++j) {
        const int e = j * 32 + lane;        // 0..127 16B-packets
        const int row = e >> 2, ch = e & 3; // kv row, 16B chunk
        const __bf16* g = KH + (size_t)(bb * NKV + kv0n + row) * CC + h * HD + ch * 8;
        async_lds_b128((unsigned)(size_t)&Klds[buf][row * LROW + ch * 8], g);
      }
    } else if (wib == 1) {
      #pragma unroll
      for (int j = 0; j < 4; ++j) {
        const int e = j * 32 + lane;
        const int row = e >> 2, ch = e & 3; // d row, 16B chunk
        const __bf16* g = VHT + (size_t)(bb * CC + h * HD + row) * NKV + kv0n + ch * 8;
        async_lds_b128((unsigned)(size_t)&Vlds[buf][row * LROW + ch * 8], g);
      }
    }
  };

  issue_chunk(0, 0);
  if (wib < 2) wait_async0();
  __syncthreads();

  for (int c = 0; c < NKV / 32; ++c) {
    const int buf = c & 1;
    const int kv0 = c * 32;
    if (c + 1 < NKV / 32) issue_chunk(buf ^ 1, kv0 + 32);

    // score fragments from LDS (K: row=kv_local, col=d)
    v16bf kf0 = ld_bf16x16(&Klds[buf][(lo) * LROW + hi * 16]);
    v16bf kf1 = ld_bf16x16(&Klds[buf][(16 + lo) * LROW + hi * 16]);

    v8f zero = {};
    v8f s0 = WMMA_BF16(qf, kf0, zero);
    v8f s1 = WMMA_BF16(qf, kf1, zero);

    // mask + online softmax (reduce across 16-lane halves; wave32 shfl)
    #pragma unroll
    for (int vv = 0; vv < 8; ++vv) {
      float a = s0[vv];
      float csc = s1[vv];
      if (mbase[(size_t)vv * NKV + kv0])      a   = -1e30f;  // mask true -> -inf
      if (mbase[(size_t)vv * NKV + kv0 + 16]) csc = -1e30f;
      float mx = fmaxf(a, csc);
      mx = fmaxf(mx, __shfl_xor(mx, 1, 32));
      mx = fmaxf(mx, __shfl_xor(mx, 2, 32));
      mx = fmaxf(mx, __shfl_xor(mx, 4, 32));
      mx = fmaxf(mx, __shfl_xor(mx, 8, 32));
      const float newm = fmaxf(run_m[vv], mx);
      const float pa = (a   <= -1e29f) ? 0.0f : __expf(a - newm);
      const float pc = (csc <= -1e29f) ? 0.0f : __expf(csc - newm);
      const float corr = __expf(run_m[vv] - newm);
      float ps = pa + pc;
      ps += __shfl_xor(ps, 1, 32);
      ps += __shfl_xor(ps, 2, 32);
      ps += __shfl_xor(ps, 4, 32);
      ps += __shfl_xor(ps, 8, 32);
      run_s[vv] = run_s[vv] * corr + ps;
      run_m[vv] = newm;
      o0[vv] *= corr;
      o1[vv] *= corr;
      // stage P (D layout row = vv + 8*hi, cols lane%16 / +16) into LDS
      Plds[wib][(vv + 8 * hi) * 32 + lo]      = (__bf16)pa;
      Plds[wib][(vv + 8 * hi) * 32 + 16 + lo] = (__bf16)pc;
    }

    // reload P in A-fragment layout (within-wave DS ops are in-order)
    const __bf16* pr = &Plds[wib][lo * 32 + hi * 16];
    v16bf pf = ld_bf16x16(pr);

    // V fragments from LDS (V: row=d, col=kv_local)
    v16bf vf0 = ld_bf16x16(&Vlds[buf][(lo) * LROW + hi * 16]);
    v16bf vf1 = ld_bf16x16(&Vlds[buf][(16 + lo) * LROW + hi * 16]);

    o0 = WMMA_BF16(pf, vf0, o0);
    o1 = WMMA_BF16(pf, vf1, o1);

    if (wib < 2) wait_async0();   // next chunk landed before publishing
    __syncthreads();
  }

  // normalize and store X (B, NQ, C) f32
  #pragma unroll
  for (int vv = 0; vv < 8; ++vv) {
    const float inv = 1.0f / run_s[vv];
    const int qrow_i = qBase + vv + 8 * hi;
    float* xp = X + (size_t)(bb * NQ + qrow_i) * CC + h * HD;
    xp[lo]      = o0[vv] * inv;
    xp[16 + lo] = o1[vv] * inv;
  }
}

// ---------------------------------------------------------------------------
// Stage C: out = X @ Wp^T + bp, f32 output. Same tiling as stage A, M=1024.
// ---------------------------------------------------------------------------
__global__ __launch_bounds__(128) void OutProjKernel(
    const float* __restrict__ X, const float* __restrict__ Wp,
    const float* __restrict__ bp, float* __restrict__ out)
{
  const int lane = threadIdx.x & 31;
  const int wib  = threadIdx.x >> 5;
  const int lo   = lane & 15;
  const int hi   = lane >> 4;
  const int m0   = blockIdx.x * 16;
  const int n00  = wib * 64;

  v8f acc[4] = {};
  const float* arow = X + (size_t)(m0 + lo) * CC;
  #pragma unroll
  for (int kt = 0; kt < 8; ++kt) {
    v16bf afrag = cvt_f32x16(arow + kt * 32 + hi * 16);
    #pragma unroll
    for (int nt = 0; nt < 4; ++nt) {
      const float* wrow = Wp + (size_t)(n00 + nt * 16 + lo) * CC + kt * 32 + hi * 16;
      v16bf bfrag = cvt_f32x16(wrow);
      acc[nt] = WMMA_BF16(afrag, bfrag, acc[nt]);
    }
  }
  #pragma unroll
  for (int nt = 0; nt < 4; ++nt) {
    const int col = n00 + nt * 16 + lo;
    const float bias = bp[col];
    #pragma unroll
    for (int vv = 0; vv < 8; ++vv) {
      out[(size_t)(m0 + vv + 8 * hi) * CC + col] = acc[nt][vv] + bias;
    }
  }
}

// ---------------------------------------------------------------------------
extern "C" void kernel_launch(void* const* d_in, const int* in_sizes, int n_in,
                              void* d_out, int out_size, void* d_ws, size_t ws_size,
                              hipStream_t stream) {
  const float* q  = (const float*)d_in[0];
  const float* k  = (const float*)d_in[1];
  const float* v  = (const float*)d_in[2];
  const float* Wk = (const float*)d_in[3];
  const float* Wv = (const float*)d_in[4];
  const float* Wp = (const float*)d_in[5];
  const float* bp = (const float*)d_in[6];
  const unsigned char* mask = (const unsigned char*)d_in[7];  // jax bool = 1 byte
  float* out = (float*)d_out;

  char* ws = (char*)d_ws;
  __bf16* KH  = (__bf16*)(ws);                                  // 16 MB
  __bf16* VHT = (__bf16*)(ws + (size_t)16 * 1024 * 1024);       // 16 MB
  __bf16* QH  = (__bf16*)(ws + (size_t)32 * 1024 * 1024);       // 0.5 MB
  float*  X   = (float*) (ws + (size_t)33 * 1024 * 1024);       // 1 MB

  // Stage A: K/V/Q projections (job 2 only uses first 64 row strips)
  ProjKernel<<<dim3(2048, 3, 1), 128, 0, stream>>>(q, k, v, Wk, Wv, KH, VHT, QH);
  // Stage B: flash attention, one block per (b,h), async K/V staging
  AttnKernel<<<dim3(32, 1, 1), 512, 0, stream>>>(QH, KH, VHT, mask, X);
  // Stage C: output projection + bias
  OutProjKernel<<<dim3(64, 1, 1), 128, 0, stream>>>(X, Wp, bp, out);
}
